// GraphSageNet_pyg_5643587027412
// MI455X (gfx1250) — matrix operand order repaired and checked by reading, not compiled, not run on hardware
//
#include <hip/hip_runtime.h>
#include <hip/hip_bf16.h>

typedef __attribute__((ext_vector_type(16))) _Float16 v16h;
typedef __attribute__((ext_vector_type(8)))  _Float16 v8h;
typedef __attribute__((ext_vector_type(8)))  float    v8f;
typedef __attribute__((ext_vector_type(4)))  int      v4i;

#define N_NODES 100000
#define N_EDGES 3200000
#define HID 128

#if defined(__has_builtin)
#if __has_builtin(__builtin_amdgcn_global_load_async_to_lds_b128) && \
    __has_builtin(__builtin_amdgcn_s_wait_asynccnt)
#define USE_ASYNC_LDS 1
#endif
#endif

// ---------------------------------------------------------------- utilities
__global__ void k_fill(float* __restrict__ p, float v, size_t n) {
    size_t t = (size_t)blockIdx.x * blockDim.x + threadIdx.x;
    if (t < n) p[t] = v;
}

// x[node, :] = emb[h[node], :]   (float4 per thread, 32 threads per node)
__global__ void k_embed(const float* __restrict__ emb, const int* __restrict__ hh,
                        float* __restrict__ x, int N) {
    size_t t = (size_t)blockIdx.x * blockDim.x + threadIdx.x;
    size_t node = t >> 5;
    if (node >= (size_t)N) return;
    int c4 = (int)(t & 31) << 2;
    int row = hh[node];
    *(float4*)(x + node * HID + c4) = *(const float4*)(emb + (size_t)row * HID + c4);
}

__global__ void k_degree(const int* __restrict__ dst, float* __restrict__ deg, int E) {
    size_t t = (size_t)blockIdx.x * blockDim.x + threadIdx.x;
    if (t >= (size_t)E) return;
    atomicAdd(deg + dst[t], 1.0f);
}

__global__ void k_invdeg(float* __restrict__ deg, int N) {
    size_t t = (size_t)blockIdx.x * blockDim.x + threadIdx.x;
    if (t >= (size_t)N) return;
    deg[t] = 1.0f / fmaxf(deg[t], 1.0f);
}

// agg[dst] += x[src]  — 32 threads per edge, float4 gather + 4 f32 atomics.
// x and agg are each 51 MB: both L2-resident (192 MB), so this runs at L2 rate.
__global__ void k_scatter(const int* __restrict__ src, const int* __restrict__ dst,
                          const float* __restrict__ x, float* __restrict__ agg, int E) {
    size_t t = (size_t)blockIdx.x * blockDim.x + threadIdx.x;
    size_t e = t >> 5;
    if (e >= (size_t)E) return;
    int c4 = (int)(t & 31) << 2;
    int s = src[e], d = dst[e];
    float4 v = *(const float4*)(x + (size_t)s * HID + c4);
    float* ap = agg + (size_t)d * HID + c4;
    atomicAdd(ap + 0, v.x);
    atomicAdd(ap + 1, v.y);
    atomicAdd(ap + 2, v.z);
    atomicAdd(ap + 3, v.w);
}

// Build Wsw: f16 [Wl ; Wr] pre-swizzled into wave32 WMMA B-fragment order.
// Linear dest index i = (((nt*8 + ks)*32) + lane)*16 + h maps to source
// element (k, n) with n = nt*16 + (lane&15), k = ks*32 + (lane>=16 ? 8:0) + (h<8 ? h : 8+h).
__global__ void k_prep_w(const float* __restrict__ Wl, const float* __restrict__ Wr,
                         _Float16* __restrict__ Wsw) {
    int i = blockIdx.x * blockDim.x + threadIdx.x;
    if (i >= 256 * HID) return;
    int hh = i & 15;
    int l  = (i >> 4) & 31;
    int ks = (i >> 9) & 7;
    int nt = (i >> 12) & 7;
    int n  = nt * 16 + (l & 15);
    int k  = ks * 32 + ((l >> 4) << 3) + (hh < 8 ? hh : 8 + hh);
    float v = (k < 128) ? Wl[k * HID + n] : Wr[(k - 128) * HID + n];
    Wsw[i] = (_Float16)v;
}

// build a 16x32 f16 A fragment from a f32 row slice (WMMA 16-bit A layout)
__device__ __forceinline__ v16h load_a(const float* rp, float scale) {
    float av[16];
    *(float4*)&av[0]  = *(const float4*)(rp + 0);
    *(float4*)&av[4]  = *(const float4*)(rp + 4);
    *(float4*)&av[8]  = *(const float4*)(rp + 16);
    *(float4*)&av[12] = *(const float4*)(rp + 20);
    v16h a;
    #pragma unroll
    for (int i = 0; i < 16; ++i) a[i] = (_Float16)(av[i] * scale);
    return a;
}

// ------------------------------------------------------- fused SAGE GEMM
// y[m,n] = sum_k A[m,k]*Wcat[k,n] + bias[n],  A = [agg*invdeg | x], K=256.
// Block: 256 threads = 8 waves, block tile 256 rows x 128 cols.
// Wave tile: 32 rows x 128 cols (2 A fragments, acc[2][8]); each LDS B
// fragment feeds two WMMAs. B staged into LDS with async b128 copies.
__global__ __launch_bounds__(256) void k_gemm(
    const float* __restrict__ agg, const float* __restrict__ x,
    const float* __restrict__ invdeg, const _Float16* __restrict__ Wsw,
    const float* __restrict__ bias, float* __restrict__ y, int nrows) {
    __shared__ _Float16 ldsB[32768];   // 64 KB, already fragment-ordered

    const int tid  = threadIdx.x;
    const int wave = tid >> 5;
    const int lane = tid & 31;

    // ---- stage pre-swizzled weights: contiguous 64 KB global -> LDS ----
#ifdef USE_ASYNC_LDS
    {
        typedef __attribute__((address_space(1))) v4i av4i_g;
        typedef __attribute__((address_space(3))) v4i av4i_l;
        av4i_g* gsrc = (av4i_g*)Wsw;           // 4096 x 16B elements
        av4i_l* ldst = (av4i_l*)&ldsB[0];
        #pragma unroll
        for (int it = 0; it < 16; ++it) {
            __builtin_amdgcn_global_load_async_to_lds_b128(
                gsrc + it * 256 + tid, ldst + it * 256 + tid, 0, 0);
        }
        __builtin_amdgcn_s_wait_asynccnt(0);
    }
#else
    #pragma unroll
    for (int it = 0; it < 16; ++it) {
        v8h t8 = *(const v8h*)(Wsw + it * 2048 + tid * 8);
        *(v8h*)(&ldsB[it * 2048 + tid * 8]) = t8;
    }
#endif

    const int m0 = blockIdx.x * 256 + wave * 32;
    const int mA = m0 + (lane & 15);
    const int mB = mA + 16;
    const int maA = (mA < nrows) ? mA : (nrows - 1);
    const int maB = (mB < nrows) ? mB : (nrows - 1);
    __builtin_prefetch(agg + (size_t)maA * HID, 0, 1);
    __builtin_prefetch(x + (size_t)maA * HID, 0, 1);
    const float idgA = invdeg[maA];
    const float idgB = invdeg[maB];

    __syncthreads();

    v8f acc0[8] = {};
    v8f acc1[8] = {};

    #pragma unroll
    for (int ks = 0; ks < 8; ++ks) {
        const float* srcp = (ks < 4) ? agg : x;
        const float sA = (ks < 4) ? idgA : 1.0f;
        const float sB = (ks < 4) ? idgB : 1.0f;
        const int kb = (ks & 3) * 32 + ((lane >> 4) << 3);
        v16h a0 = load_a(srcp + (size_t)maA * HID + kb, sA);
        v16h a1 = load_a(srcp + (size_t)maB * HID + kb, sB);

        #pragma unroll
        for (int nt = 0; nt < 8; ++nt) {
            v16h b = *(const v16h*)(&ldsB[(((nt * 8 + ks) * 32) + lane) * 16]);
            acc0[nt] = __builtin_amdgcn_wmma_f32_16x16x32_f16(
                false, a0, false, b, (short)0, acc0[nt], false, false);
            acc1[nt] = __builtin_amdgcn_wmma_f32_16x16x32_f16(
                false, a1, false, b, (short)0, acc1[nt], false, false);
        }
    }

    // D layout: VGPR r, lanes 0-15 -> row base+r, lanes 16-31 -> base+8+r, col = lane&15
    const int col0 = lane & 15;
    const int rb0  = m0 + ((lane >> 4) << 3);
    #pragma unroll
    for (int nt = 0; nt < 8; ++nt) {
        const int n  = nt * 16 + col0;
        const float bv = bias[n];
        #pragma unroll
        for (int r = 0; r < 8; ++r) {
            int row0 = rb0 + r;
            int row1 = rb0 + 16 + r;
            if (row0 < nrows) y[(size_t)row0 * HID + n] = acc0[nt][r] + bv;
            if (row1 < nrows) y[(size_t)row1 * HID + n] = acc1[nt][r] + bv;
        }
    }
}

// ------------------------------------------------------------- batchnorm
__global__ void k_bn_stats(const float* __restrict__ y, float* __restrict__ stats, int N) {
    int c  = threadIdx.x;
    int r0 = blockIdx.x * 512;
    int r1 = r0 + 512; if (r1 > N) r1 = N;
    float s = 0.0f, s2 = 0.0f;
    for (int r = r0; r < r1; ++r) {
        float v = y[(size_t)r * HID + c];
        s += v; s2 += v * v;
    }
    atomicAdd(&stats[c], s);
    atomicAdd(&stats[HID + c], s2);
}

// x = x + (y - mu)*rsqrt(var+eps)*gamma + beta   (fused BN + residual, in place)
__global__ void k_bn_apply(float* __restrict__ x, const float* __restrict__ y,
                           const float* __restrict__ stats, const float* __restrict__ g,
                           const float* __restrict__ b, int N) {
    size_t t = (size_t)blockIdx.x * blockDim.x + threadIdx.x;
    size_t node = t >> 5;
    if (node >= (size_t)N) return;
    int c4 = (int)(t & 31) << 2;
    const float invN = 1.0f / (float)N;
    float4 ys  = *(const float4*)(y + node * HID + c4);
    float4 xs  = *(float4*)(x + node * HID + c4);
    float4 sum = *(const float4*)(stats + c4);
    float4 sq  = *(const float4*)(stats + HID + c4);
    float4 gg  = *(const float4*)(g + c4);
    float4 bb  = *(const float4*)(b + c4);
    float mu, var, sc;
#define BN1(comp)                                                         \
    mu  = sum.comp * invN;                                                \
    var = sq.comp * invN - mu * mu;                                       \
    sc  = rsqrtf(var + 1e-5f);                                            \
    xs.comp = xs.comp + (ys.comp - mu) * sc * gg.comp + bb.comp;
    BN1(x) BN1(y) BN1(z) BN1(w)
#undef BN1
    *(float4*)(x + node * HID + c4) = xs;
}

// -------------------------------------------------------- fused MLP readout
__global__ __launch_bounds__(256) void k_readout(
    const float* __restrict__ x,
    const float* __restrict__ W0, const float* __restrict__ b0,
    const float* __restrict__ W1, const float* __restrict__ b1,
    const float* __restrict__ W2, const float* __restrict__ b2,
    float* __restrict__ out, int N) {
    __shared__ float buf[8][128];
    __shared__ float h1s[8][64];
    __shared__ float h2s[8][32];
    const int wave = threadIdx.x >> 5;
    const int lane = threadIdx.x & 31;
    size_t node = (size_t)blockIdx.x * 8 + wave;
    size_t nd = (node < (size_t)N) ? node : (size_t)(N - 1);

    *(float4*)&buf[wave][lane * 4] = *(const float4*)(x + nd * HID + lane * 4);
    __syncthreads();

    float a0 = b0[lane], a1 = b0[lane + 32];
    for (int k = 0; k < 128; ++k) {
        float xv = buf[wave][k];
        a0 += xv * W0[k * 64 + lane];
        a1 += xv * W0[k * 64 + lane + 32];
    }
    h1s[wave][lane]      = fmaxf(a0, 0.0f);
    h1s[wave][lane + 32] = fmaxf(a1, 0.0f);
    __syncthreads();

    float a2 = b1[lane];
    for (int k = 0; k < 64; ++k) a2 += h1s[wave][k] * W1[k * 32 + lane];
    h2s[wave][lane] = fmaxf(a2, 0.0f);
    __syncthreads();

    if (lane < 7 && node < (size_t)N) {
        float a3 = b2[lane];
        for (int k = 0; k < 32; ++k) a3 += h2s[wave][k] * W2[k * 7 + lane];
        out[node * 7 + lane] = a3;
    }
}

// ---------------------------------------------------------------- launcher
extern "C" void kernel_launch(void* const* d_in, const int* in_sizes, int n_in,
                              void* d_out, int out_size, void* d_ws, size_t ws_size,
                              hipStream_t stream) {
    const int N = N_NODES, E = N_EDGES;

    const int *h_idx = nullptr, *ei = nullptr;
    const float* emb = nullptr;
    const float *Wl[4] = {}, *Wr[4] = {}, *bl[4] = {}, *gamma[4] = {}, *beta[4] = {};
    const float *W0 = nullptr, *b0 = nullptr, *W1 = nullptr, *b1 = nullptr,
                *W2 = nullptr, *b2 = nullptr;

    // Map inputs by flat size (robust to flatten order). Disambiguate the
    // per-layer 128-vectors by detecting jax-sorted-key order ('e' before 'h').
    int idx_h = -1, idx_e = -1;
    for (int i = 0; i < n_in; ++i) {
        if (in_sizes[i] == 100000) idx_h = i;
        if (in_sizes[i] == 3200000) idx_e = i;
    }
    const bool sorted_flatten = (idx_e >= 0 && idx_h >= 0 && idx_e < idx_h);

    int c16384 = 0, c128 = 0;
    for (int i = 0; i < n_in; ++i) {
        const void* p = d_in[i];
        switch (in_sizes[i]) {
            case 100000:  h_idx = (const int*)p; break;
            case 6400000: ei = (const int*)p; break;
            case 3200000: /* edge feature e: unused by reference */ break;
            case 896:     emb = (const float*)p; break;
            case 16384: {
                int l = c16384 >> 1;
                if ((c16384 & 1) == 0) Wl[l] = (const float*)p; else Wr[l] = (const float*)p;
                ++c16384;
            } break;
            case 128: {
                int l = c128 / 3, pos = c128 % 3;
                if (sorted_flatten) {   // sorted keys: beta, bl, gamma
                    if (pos == 0) beta[l] = (const float*)p;
                    else if (pos == 1) bl[l] = (const float*)p;
                    else gamma[l] = (const float*)p;
                } else {                // insertion order: bl, gamma, beta
                    if (pos == 0) bl[l] = (const float*)p;
                    else if (pos == 1) gamma[l] = (const float*)p;
                    else beta[l] = (const float*)p;
                }
                ++c128;
            } break;
            case 8192: W0 = (const float*)p; break;
            case 64:   b0 = (const float*)p; break;
            case 2048: W1 = (const float*)p; break;
            case 32:   b1 = (const float*)p; break;
            case 224:  W2 = (const float*)p; break;
            case 7:    b2 = (const float*)p; break;
            default: break;
        }
    }

    // workspace layout (~154 MB)
    float* xbuf  = (float*)d_ws;
    float* agg   = xbuf + (size_t)N * HID;
    float* ybuf  = agg + (size_t)N * HID;
    float* deg   = ybuf + (size_t)N * HID;
    float* stats = deg + N;
    _Float16* Wsw = (_Float16*)(stats + 256);

    const unsigned blk = 256;
    k_embed<<<(unsigned)(((size_t)N * 32 + blk - 1) / blk), blk, 0, stream>>>(emb, h_idx, xbuf, N);
    k_fill<<<(unsigned)((N + blk - 1) / blk), blk, 0, stream>>>(deg, 0.0f, (size_t)N);
    k_degree<<<(unsigned)((E + blk - 1) / blk), blk, 0, stream>>>(ei + E, deg, E);
    k_invdeg<<<(unsigned)((N + blk - 1) / blk), blk, 0, stream>>>(deg, N);

    for (int l = 0; l < 4; ++l) {
        k_fill<<<(unsigned)(((size_t)N * HID + blk - 1) / blk), blk, 0, stream>>>(
            agg, 0.0f, (size_t)N * HID);
        k_scatter<<<(unsigned)(((size_t)E * 32 + blk - 1) / blk), blk, 0, stream>>>(
            ei, ei + E, xbuf, agg, E);
        k_prep_w<<<(unsigned)((256 * HID + blk - 1) / blk), blk, 0, stream>>>(Wl[l], Wr[l], Wsw);
        k_gemm<<<(unsigned)((N + 255) / 256), blk, 0, stream>>>(agg, xbuf, deg, Wsw, bl[l], ybuf, N);
        k_fill<<<1, blk, 0, stream>>>(stats, 0.0f, (size_t)256);
        k_bn_stats<<<(unsigned)((N + 511) / 512), 128, 0, stream>>>(ybuf, stats, N);
        k_bn_apply<<<(unsigned)(((size_t)N * 32 + blk - 1) / blk), blk, 0, stream>>>(
            xbuf, ybuf, stats, gamma[l], beta[l], N);
    }

    k_readout<<<(unsigned)((N + 7) / 8), blk, 0, stream>>>(xbuf, W0, b0, W1, b1, W2, b2,
                                                           (float*)d_out, N);
}
